// Attention_7584912245222
// MI455X (gfx1250) — 8-bit matrix operand order repaired and checked by reading, compiled, not run on hardware
//
#include <hip/hip_runtime.h>


#define NB 64
#define NT 2048
#define ND 1024
#define NQ 1024

typedef int    v16i  __attribute__((ext_vector_type(16)));
typedef float  v8f   __attribute__((ext_vector_type(8)));
typedef unsigned int u32x4 __attribute__((ext_vector_type(4)));
typedef float  f32x4 __attribute__((ext_vector_type(4)));

union Frag8 { v16i v; u32x4 q[4]; };

#define WH_SCALE   16.0f
#define WH_UNSCALE 0.0625f

#define SLOT 80

#define A_BYTES (8 * 8 * 32 * SLOT)
#define B_BYTES (4 * 8 * 32 * SLOT)
#define SMEM_BYTES (A_BYTES + B_BYTES + ND * 4 + ND * 4)

__device__ __forceinline__ unsigned pack4_fp8(float a, float b, float c, float d) {
    int lo = __builtin_amdgcn_cvt_pk_fp8_f32(a, b, 0, false);
    int hi = __builtin_amdgcn_cvt_pk_fp8_f32(c, d, lo, true);
    return (unsigned)hi;
}

__device__ __forceinline__ float tanh_fast(float x) {
    float t = __expf(2.0f * x);
    return 1.0f - 2.0f * __builtin_amdgcn_rcpf(t + 1.0f);
}


__global__ void wh8_convert_kernel(const float* __restrict__ Wh,
                                   unsigned* __restrict__ WhT8) {
    const int dw = blockIdx.x * 256 + threadIdx.x;
    const int c    = dw >> 2;
    const int j0   = (dw & 3) * 4;
    const int q    = c & 3;
    const int lane = (c >> 2) & 31;
    const int kc   = (c >> 7) & 7;
    const int nt   = c >> 10;
    const int n = nt * 16 + (lane & 15);
    const int k = kc * 128 + q * 32 + ((lane >> 4) << 4) + j0;
    float f0 = WH_SCALE * Wh[(size_t)(k + 0) * ND + n];
    float f1 = WH_SCALE * Wh[(size_t)(k + 1) * ND + n];
    float f2 = WH_SCALE * Wh[(size_t)(k + 2) * ND + n];
    float f3 = WH_SCALE * Wh[(size_t)(k + 3) * ND + n];
    WhT8[dw] = pack4_fp8(f0, f1, f2, f3);
}

__global__ void qproj_kernel(const float* __restrict__ query,
                             const float* __restrict__ Wq,
                             float* __restrict__ qp) {
    const int e = blockIdx.x * 256 + threadIdx.x;
    const int b = blockIdx.y;
    float acc = 0.0f;
    const float* qrow = query + (size_t)b * NQ;
    #pragma unroll 8
    for (int q = 0; q < NQ; ++q)
        acc = __builtin_fmaf(qrow[q], Wq[(size_t)q * ND + e], acc);
    qp[(size_t)b * ND + e] = acc;
}


__global__ __launch_bounds__(256)
void attn_scores_kernel(const float* __restrict__ memory,
                        const unsigned* __restrict__ WhT8,
                        const float* __restrict__ qp,
                        const float* __restrict__ vvec,
                        float* __restrict__ s_out) {
    extern __shared__ __align__(16) unsigned char smem[];
    unsigned char* A_lds = smem;
    unsigned char* B_lds = smem + A_BYTES;
    float* v_lds = (float*)(smem + A_BYTES + B_BYTES);
    float* q_lds = v_lds + ND;

    const int tid  = threadIdx.x;
    const int lane = tid & 31;
    const int wave = tid >> 5;
    const int row0 = blockIdx.x * 128;
    const int b    = row0 >> 11;

    {
        const f32x4* src = (const f32x4*)(memory + (size_t)row0 * ND);
        #pragma unroll 4
        for (int i = 0; i < 128; ++i) {
            int cc   = tid + i * 256;
            int row  = cc >> 8;
            int kk   = (cc & 255) * 4;
            f32x4 f  = src[(size_t)row * 256 + (cc & 255)];
            unsigned packed = pack4_fp8(f.x, f.y, f.z, f.w);
            int g = row >> 4, mrow_s = row & 15;
            int kc = kk >> 7, kl = kk & 127;
            int blk = kl >> 5, klb = kl & 31;
            int p = klb >> 4, w16 = klb & 15;
            int half = p, w = w16 >> 2;
            int lane_t = mrow_s + (half << 4);
            int d = blk * 4 + w;
            unsigned dst = (unsigned)(((g * 8 + kc) * 32 + lane_t) * SLOT + d * 4);
            *(unsigned*)(A_lds + dst) = packed;
        }
        #pragma unroll
        for (int j = 0; j < 4; ++j) {
            v_lds[tid * 4 + j] = vvec[tid * 4 + j];
            q_lds[tid * 4 + j] = qp[(size_t)b * ND + tid * 4 + j];
        }
    }
    __syncthreads();

    const int mrow = lane & 15;
    const unsigned alane = (unsigned)((wave * 8 * 32 + lane) * SLOT);
    const unsigned blane = (unsigned)(lane * SLOT);

    float sacc[8];
    #pragma unroll
    for (int r = 0; r < 8; ++r) sacc[r] = 0.0f;

    for (int quad = 0; quad < 16; ++quad) {
        __syncthreads();
        #pragma unroll
        for (int i = 0; i < 16; ++i) {
            int cc = tid + i * 256;
            unsigned dst = (unsigned)((cc >> 2) * SLOT + (cc & 3) * 16);
            *(u32x4*)(B_lds + dst) =
                *(const u32x4*)(WhT8 + (size_t)quad * 16384 + (size_t)cc * 4);
        }
        __syncthreads();

        v8f acc[4];
        #pragma unroll
        for (int t = 0; t < 4; ++t)
            acc[t] = (v8f){0.f,0.f,0.f,0.f,0.f,0.f,0.f,0.f};

        Frag8 afr[2], bfr[2];
        {
            const unsigned ab = alane;
            afr[0].q[0] = *(const u32x4*)(A_lds + ab);
            afr[0].q[1] = *(const u32x4*)(A_lds + ab + 16);
            afr[0].q[2] = *(const u32x4*)(A_lds + ab + 32);
            afr[0].q[3] = *(const u32x4*)(A_lds + ab + 48);
            const unsigned bb = blane;
            bfr[0].q[0] = *(const u32x4*)(B_lds + bb);
            bfr[0].q[1] = *(const u32x4*)(B_lds + bb + 16);
            bfr[0].q[2] = *(const u32x4*)(B_lds + bb + 32);
            bfr[0].q[3] = *(const u32x4*)(B_lds + bb + 48);
        }
        #pragma unroll
        for (int kc = 0; kc < 8; ++kc) {
            #pragma unroll
            for (int t = 0; t < 4; ++t) {
                const int idx = kc * 4 + t;
                const int cur = idx & 1;
                const int nidx = idx + 1;
                if (nidx < 32) {
                    const int nkc = nidx >> 2, nt2 = nidx & 3;
                    const unsigned bb =
                        blane + (unsigned)((nt2 * 8 + nkc) * 32 * SLOT);
                    bfr[cur ^ 1].q[0] = *(const u32x4*)(B_lds + bb);
                    bfr[cur ^ 1].q[1] = *(const u32x4*)(B_lds + bb + 16);
                    bfr[cur ^ 1].q[2] = *(const u32x4*)(B_lds + bb + 32);
                    bfr[cur ^ 1].q[3] = *(const u32x4*)(B_lds + bb + 48);
                    if (nt2 == 0) {
                        const unsigned ab =
                            alane + (unsigned)(nkc * 32 * SLOT);
                        afr[nkc & 1].q[0] = *(const u32x4*)(A_lds + ab);
                        afr[nkc & 1].q[1] = *(const u32x4*)(A_lds + ab + 16);
                        afr[nkc & 1].q[2] = *(const u32x4*)(A_lds + ab + 32);
                        afr[nkc & 1].q[3] = *(const u32x4*)(A_lds + ab + 48);
                    }
                }
                acc[t] = __builtin_amdgcn_wmma_f32_16x16x128_fp8_fp8(
                             afr[kc & 1].v, bfr[cur].v, (short)0, acc[t],
                             false, false);
            }
        }

        const int nbase = quad * 64;
        #pragma unroll
        for (int t = 0; t < 4; ++t) {
            const int e = nbase + t * 16 + mrow;
            const float vv = v_lds[e];
            const float qq = q_lds[e];
            #pragma unroll
            for (int r = 0; r < 8; ++r)
                sacc[r] += tanh_fast(acc[t][r] * WH_UNSCALE + qq) * vv;
        }
    }

    #pragma unroll
    for (int off = 8; off >= 1; off >>= 1) {
        #pragma unroll
        for (int r = 0; r < 8; ++r)
            sacc[r] += __shfl_xor(sacc[r], off, 16);
    }
    if ((lane & 15) == 0) {
        const int base = row0 + wave * 16 + ((lane >> 4) << 3);
        #pragma unroll
        for (int r = 0; r < 8; ++r) s_out[base + r] = sacc[r];
    }
}

__global__ void softmax_kernel(const float* __restrict__ s,
                               float* __restrict__ p) {
    __shared__ float red[256];
    const int b = blockIdx.x, tid = threadIdx.x;
    const float* sb = s + (size_t)b * NT;
    float m = -3.4e38f;
    for (int i = tid; i < NT; i += 256) m = fmaxf(m, sb[i]);
    red[tid] = m; __syncthreads();
    for (int o = 128; o >= 1; o >>= 1) { if (tid < o) red[tid] = fmaxf(red[tid], red[tid + o]); __syncthreads(); }
    m = red[0]; __syncthreads();
    float sum = 0.0f;
    float* pb = p + (size_t)b * NT;
    for (int i = tid; i < NT; i += 256) { float e = __expf(sb[i] - m); pb[i] = e; sum += e; }
    red[tid] = sum; __syncthreads();
    for (int o = 128; o >= 1; o >>= 1) { if (tid < o) red[tid] += red[tid + o]; __syncthreads(); }
    const float inv = __builtin_amdgcn_rcpf(red[0]);
    for (int i = tid; i < NT; i += 256) pb[i] *= inv;
}

__global__ __launch_bounds__(256)
void wsum_kernel(const float* __restrict__ memory,
                 const float* __restrict__ p,
                 float* __restrict__ out) {
    __shared__ float pl[NT];
    const int b = blockIdx.y;
    const int d = blockIdx.x * 256 + threadIdx.x;
    for (int i = threadIdx.x; i < NT; i += 256) pl[i] = p[(size_t)b * NT + i];
    __syncthreads();
    float acc = 0.0f;
    const float* mb = memory + (size_t)b * NT * ND + d;
    #pragma unroll 8
    for (int t = 0; t < NT; ++t)
        acc = __builtin_fmaf(mb[(size_t)t * ND], pl[t], acc);
    out[(size_t)b * ND + d] = acc;
}

extern "C" void kernel_launch(void* const* d_in, const int* in_sizes, int n_in,
                              void* d_out, int out_size, void* d_ws, size_t ws_size,
                              hipStream_t stream) {
    const float* memory = (const float*)d_in[0];
    const float* query  = (const float*)d_in[1];
    const float* Wh     = (const float*)d_in[2];
    const float* Wq     = (const float*)d_in[3];
    const float* vvec   = (const float*)d_in[4];
    float* out = (float*)d_out;

    unsigned char* ws = (unsigned char*)d_ws;
    unsigned* WhT8 = (unsigned*)ws;
    float* qp = (float*)(ws + (size_t)ND * ND);
    float* s  = (float*)((unsigned char*)qp + (size_t)NB * ND * 4);
    float* p  = (float*)((unsigned char*)s  + (size_t)NB * NT * 4);

    (void)hipFuncSetAttribute((const void*)attn_scores_kernel,
                              hipFuncAttributeMaxDynamicSharedMemorySize,
                              SMEM_BYTES);

    wh8_convert_kernel<<<dim3((ND * ND / 4) / 256), 256, 0, stream>>>(Wh, WhT8);
    qproj_kernel<<<dim3(ND / 256, NB), 256, 0, stream>>>(query, Wq, qp);
    attn_scores_kernel<<<dim3((NB * NT) / 128), 256, SMEM_BYTES, stream>>>(
        memory, WhT8, qp, vvec, s);
    softmax_kernel<<<dim3(NB), 256, 0, stream>>>(s, p);
    wsum_kernel<<<dim3(ND / 256, NB), 256, 0, stream>>>(memory, p, out);
}
